// DeepRecursiveNetwork_46084999086231
// MI455X (gfx1250) — compile-verified, matrix operand order checked by reading, not run on hardware
//
#include <hip/hip_runtime.h>
#include <math.h>

// Problem constants (match reference).
#define B_DIM    1024
#define DIN_DIM  512
#define H_DIM    1024
#define DOUT_DIM 512
#define NB_DIM   10
#define N_STEPS  30
#define N_INNER  5

typedef __bf16 bf16;
typedef __attribute__((ext_vector_type(16))) __bf16 v16bf;
typedef __attribute__((ext_vector_type(8)))  __bf16 v8bf;
typedef __attribute__((ext_vector_type(8)))  float  v8f;

union FragBF {
  v16bf v;
  v8bf  h[2];
};

// Load one 16x32 bf16 WMMA operand fragment. `off` is a 32-bit element offset
// (lane-dependent part precomputed by caller) so the compiler can emit
// saddr + 32-bit voffset addressing instead of 64-bit VGPR address pairs.
__device__ __forceinline__ v16bf load_frag(const bf16* __restrict__ base, int off) {
  FragBF f;
  f.h[0] = *(const v8bf*)(base + off);
  f.h[1] = *(const v8bf*)(base + off + 16);
  return f.v;
}

// 64x64 wave tile: 4x4 grid of 16x16 f32 accumulators.
struct Acc { v8f t[4][4]; };

// Wave-level 64x64 GEMM tile: C += A(64xK) * W(64xK)^T, bf16 in / f32 acc.
// 8 fragment loads : 16 WMMAs per K-step; per-lane offsets are loop-invariant,
// only the scalar kb advances.
__device__ __forceinline__ void gemm_tile(const bf16* __restrict__ A,
                                          const bf16* __restrict__ W,
                                          int K, int m0, int n0, Acc& acc) {
  const int lane = threadIdx.x & 31;
  const int r    = lane & 15;            // row within 16-row fragment
  const int kk   = (lane >> 4) << 3;     // K-half select: 0 or 8
  int aoff[4], woff[4];
#pragma unroll
  for (int i = 0; i < 4; ++i) aoff[i] = (m0 + 16 * i + r) * K + kk;
#pragma unroll
  for (int j = 0; j < 4; ++j) woff[j] = (n0 + 16 * j + r) * K + kk;

  for (int kb = 0; kb < K; kb += 32) {
    v16bf af[4], bfrag[4];
#pragma unroll
    for (int i = 0; i < 4; ++i) af[i] = load_frag(A, aoff[i] + kb);
#pragma unroll
    for (int j = 0; j < 4; ++j) bfrag[j] = load_frag(W, woff[j] + kb);
#pragma unroll
    for (int i = 0; i < 4; ++i)
#pragma unroll
      for (int j = 0; j < 4; ++j)
        acc.t[i][j] = __builtin_amdgcn_wmma_f32_16x16x32_bf16(
            false, af[i], false, bfrag[j], (short)0, acc.t[i][j], false, false);
  }
}

// 128 threads = 4 waves in a 2x2 grid -> 128x128 workgroup C tile.
__device__ __forceinline__ void wave_origin(int& m0, int& n0) {
  const int wave = threadIdx.x >> 5;
  m0 = blockIdx.y * 128 + (wave & 1) * 64;
  n0 = blockIdx.x * 128 + (wave >> 1) * 64;
}

// ---------------- GEMM + bias (embed / head) ----------------
__global__ __launch_bounds__(128)
void gemm_bias_kernel(const bf16* __restrict__ A,     // [M x K]
                      const bf16* __restrict__ W,     // [N x K] (row n = B col n)
                      const float* __restrict__ bias, // [N]
                      float* __restrict__ Cf,         // [M x N]
                      int M, int N, int K) {
  int m0, n0;
  wave_origin(m0, n0);
  Acc acc{};
  gemm_tile(A, W, K, m0, n0, acc);

  const int lane = threadIdx.x & 31;
  const int col  = lane & 15;
  const int rhi  = (lane >> 4) << 3;
#pragma unroll
  for (int i = 0; i < 4; ++i)
#pragma unroll
    for (int j = 0; j < 4; ++j) {
      const int gn = n0 + j * 16 + col;
      const float bv = bias[gn];
      int idx = (m0 + i * 16 + rhi) * N + gn;
#pragma unroll
      for (int p = 0; p < 8; ++p) {
        Cf[idx] = acc.t[i][j][p] + bv;
        idx += N;
      }
    }
}

// ---------------- fused inner fixed-point iteration ----------------
// hi_out[n] = 0.5*hi_in[n] + 0.5*tanh(hi_in[n] @ W[n]^T + b[n] + h[n] + prev[n])
__global__ __launch_bounds__(128)
void inner_iter_kernel(const bf16* __restrict__ hi_in,
                       bf16* __restrict__ hi_out,
                       const bf16* __restrict__ Wbf,
                       const float* __restrict__ block_b,
                       const float* __restrict__ h_all,
                       const float* __restrict__ xemb) {
  const int n = blockIdx.z;
  const size_t BH = (size_t)B_DIM * H_DIM;
  const bf16*  A    = hi_in + n * BH;
  const bf16*  W    = Wbf + (size_t)n * H_DIM * H_DIM;
  const float* bias = block_b + n * H_DIM;
  const float* hn   = h_all + n * BH;
  const float* prev = (n == 0) ? xemb : (h_all + (size_t)(n - 1) * BH);
  bf16* outp = hi_out + n * BH;

  int m0, n0;
  wave_origin(m0, n0);
  Acc acc{};
  gemm_tile(A, W, H_DIM, m0, n0, acc);

  const int lane = threadIdx.x & 31;
  const int col  = lane & 15;
  const int rhi  = (lane >> 4) << 3;
#pragma unroll
  for (int i = 0; i < 4; ++i)
#pragma unroll
    for (int j = 0; j < 4; ++j) {
      const int gn = n0 + j * 16 + col;
      const float bv = bias[gn];
      int idx = (m0 + i * 16 + rhi) * H_DIM + gn;
#pragma unroll
      for (int p = 0; p < 8; ++p) {
        const float pre = acc.t[i][j][p] + bv + hn[idx] + prev[idx];
        const float old = (float)A[idx];
        outp[idx] = (bf16)(0.5f * old + 0.5f * tanhf(pre));
        idx += H_DIM;
      }
    }
}

// ---------------- elementwise helpers ----------------
__global__ void f2bf_kernel(const float* __restrict__ in, bf16* __restrict__ out, size_t n) {
  size_t i = (size_t)blockIdx.x * blockDim.x + threadIdx.x;
  if (i < n) out[i] = (bf16)in[i];
}

__global__ void zero_kernel(float* __restrict__ p, size_t n) {
  size_t i = (size_t)blockIdx.x * blockDim.x + threadIdx.x;
  if (i < n) p[i] = 0.0f;
}

// First inner iteration from h_i == 0: h_i = 0.5*tanh(b + h + prev)
__global__ void iter0_kernel(const float* __restrict__ block_b,
                             const float* __restrict__ h_all,
                             const float* __restrict__ xemb,
                             bf16* __restrict__ hi_out) {
  size_t gid = (size_t)blockIdx.x * blockDim.x + threadIdx.x;
  const size_t BH = (size_t)B_DIM * H_DIM;
  if (gid >= (size_t)NB_DIM * BH) return;
  const int n = (int)(gid / BH);
  const size_t rem = gid - (size_t)n * BH;
  const int d = (int)(rem % H_DIM);
  const float prevv = (n == 0) ? xemb[rem] : h_all[(size_t)(n - 1) * BH + rem];
  const float pre = block_b[(size_t)n * H_DIM + d] + h_all[gid] + prevv;
  hi_out[gid] = (bf16)(0.5f * tanhf(pre));
}

// h = 0.5*h + 0.5*h_i_final
__global__ void combine_kernel(float* __restrict__ h_all, const bf16* __restrict__ hi, size_t n) {
  size_t i = (size_t)blockIdx.x * blockDim.x + threadIdx.x;
  if (i < n) h_all[i] = 0.5f * h_all[i] + 0.5f * (float)hi[i];
}

// ---------------- host orchestration ----------------
static inline char* carve(char*& p, size_t bytes) {
  char* r = p;
  p += (bytes + 255) & ~(size_t)255;
  return r;
}

extern "C" void kernel_launch(void* const* d_in, const int* /*in_sizes*/, int /*n_in*/,
                              void* d_out, int /*out_size*/, void* d_ws, size_t /*ws_size*/,
                              hipStream_t stream) {
  const float* x       = (const float*)d_in[0];
  const float* embed_W = (const float*)d_in[1];
  const float* embed_b = (const float*)d_in[2];
  const float* block_W = (const float*)d_in[3];
  const float* block_b = (const float*)d_in[4];
  const float* head_W  = (const float*)d_in[5];
  const float* head_b  = (const float*)d_in[6];
  float* out = (float*)d_out;

  const size_t BH  = (size_t)B_DIM * H_DIM;            // 1M
  const size_t NBH = (size_t)NB_DIM * BH;              // 10M

  char* p = (char*)d_ws;
  bf16*  Wbf   = (bf16*)carve(p, NBH * sizeof(bf16));                    // 20 MB
  bf16*  eWbf  = (bf16*)carve(p, (size_t)H_DIM * DIN_DIM * sizeof(bf16));// 1 MB
  bf16*  hWbf  = (bf16*)carve(p, (size_t)DOUT_DIM * H_DIM * sizeof(bf16));
  bf16*  xbf   = (bf16*)carve(p, (size_t)B_DIM * DIN_DIM * sizeof(bf16));
  float* xemb  = (float*)carve(p, BH * sizeof(float));                   // 4 MB
  float* h     = (float*)carve(p, NBH * sizeof(float));                  // 40 MB
  bf16*  hiA   = (bf16*)carve(p, NBH * sizeof(bf16));                    // 20 MB
  bf16*  hiB   = (bf16*)carve(p, NBH * sizeof(bf16));                    // 20 MB
  bf16*  h9bf  = (bf16*)carve(p, BH * sizeof(bf16));                     // 2 MB

  const int T = 256;
  auto blocks = [](size_t n) { return (unsigned)((n + 255) / 256); };

  // bf16 conversions of static operands
  f2bf_kernel<<<blocks(NBH), T, 0, stream>>>(block_W, Wbf, NBH);
  f2bf_kernel<<<blocks((size_t)H_DIM * DIN_DIM), T, 0, stream>>>(embed_W, eWbf, (size_t)H_DIM * DIN_DIM);
  f2bf_kernel<<<blocks((size_t)DOUT_DIM * H_DIM), T, 0, stream>>>(head_W, hWbf, (size_t)DOUT_DIM * H_DIM);
  f2bf_kernel<<<blocks((size_t)B_DIM * DIN_DIM), T, 0, stream>>>(x, xbf, (size_t)B_DIM * DIN_DIM);

  // x_emb = x @ embed_W^T + embed_b   (M=1024, N=1024, K=512)
  gemm_bias_kernel<<<dim3(H_DIM / 128, B_DIM / 128, 1), 128, 0, stream>>>(
      xbf, eWbf, embed_b, xemb, B_DIM, H_DIM, DIN_DIM);

  zero_kernel<<<blocks(NBH), T, 0, stream>>>(h, NBH);

  for (int t = 0; t < N_STEPS; ++t) {
    // inner iteration 0 (h_i starts at zero -> pure elementwise)
    iter0_kernel<<<blocks(NBH), T, 0, stream>>>(block_b, h, xemb, hiA);
    bf16* cur = hiA;
    bf16* nxt = hiB;
    for (int it = 1; it < N_INNER; ++it) {
      inner_iter_kernel<<<dim3(H_DIM / 128, B_DIM / 128, NB_DIM), 128, 0, stream>>>(
          cur, nxt, Wbf, block_b, h, xemb);
      bf16* tmp = cur; cur = nxt; nxt = tmp;
    }
    combine_kernel<<<blocks(NBH), T, 0, stream>>>(h, cur, NBH);
  }

  // head: out = h[NB-1] @ head_W^T + head_b   (M=1024, N=512, K=1024)
  f2bf_kernel<<<blocks(BH), T, 0, stream>>>(h + (size_t)(NB_DIM - 1) * BH, h9bf, BH);
  gemm_bias_kernel<<<dim3(DOUT_DIM / 128, B_DIM / 128, 1), 128, 0, stream>>>(
      h9bf, hWbf, head_b, out, B_DIM, DOUT_DIM, H_DIM);
}
